// Attention_25701084299349
// MI455X (gfx1250) — compile-verified, matrix operand order checked by reading, not run on hardware
//
#include <hip/hip_runtime.h>
#include <stdint.h>

// ---------------------------------------------------------------------------
// CDNA5 (gfx1250, wave32) flash-attention implementation of the reference
// non-local attention block.  bf16 WMMA (v_wmma_f32_16x16x32_bf16) with f32
// accumulation; energy matrix never materialized.  K/V blocks staged through
// LDS with async-to-LDS DMA (ASYNCcnt) and double buffering.
// ---------------------------------------------------------------------------

#define BB   4
#define CC   256
#define CQK  64
#define HW   4096   // 64*64

#if defined(__has_builtin)
#  if __has_builtin(__builtin_amdgcn_global_load_async_to_lds_b128)
#    define HAVE_ASYNC_LDS 1
#  endif
#endif
#ifndef HAVE_ASYNC_LDS
#  define HAVE_ASYNC_LDS 0
#endif

typedef __attribute__((ext_vector_type(16))) __bf16 v16bf;
typedef __attribute__((ext_vector_type(8)))  float  v8f;
typedef __attribute__((ext_vector_type(4)))  int    i32x4;

#if HAVE_ASYNC_LDS
typedef __attribute__((address_space(1))) i32x4 g_i32x4;   // global
typedef __attribute__((address_space(3))) i32x4 l_i32x4;   // LDS
#endif

union BFrag {
  v16bf v;
  unsigned short u[16];
  uint32_t d[8];
  uint4 q[2];
};

__device__ __forceinline__ unsigned short f2bf(float f) {
  union { float f; uint32_t u; } x{f};
  uint32_t r = x.u + 0x7FFFu + ((x.u >> 16) & 1u);   // round-to-nearest-even
  return (unsigned short)(r >> 16);
}

// 16-byte global -> LDS copy.  Async DMA (ASYNCcnt) when the toolchain has
// the gfx1250 builtin, otherwise load/store through VGPRs.
__device__ __forceinline__ void cp16(const unsigned short* g, unsigned short* l) {
#if HAVE_ASYNC_LDS
  __builtin_amdgcn_global_load_async_to_lds_b128(
      (g_i32x4*)g, (l_i32x4*)l, 0, 0);
#else
  *(uint4*)l = *(const uint4*)g;
#endif
}

__device__ __forceinline__ void wait_async() {
#if HAVE_ASYNC_LDS
#  if __has_builtin(__builtin_amdgcn_s_wait_asynccnt)
  __builtin_amdgcn_s_wait_asynccnt(0);
#  else
  asm volatile("s_wait_asynccnt 0" ::: "memory");
#  endif
#endif
}

// Per-lane A-layout fragment load for 16x32 bf16 (ISA 7.12.2):
//   VGPR0-3: K = half*8 + 0..7 ; VGPR4-7: K = 16 + half*8 + 0..7
// (half = lane>>4).  A B-fragment is the A-fragment of B^T, so the same
// gather works for K^T / V^T rows stored row-major.
__device__ __forceinline__ v16bf ld_frag(const unsigned short* row, int half) {
  BFrag r;
  r.q[0] = *(const uint4*)(row + half * 8);
  r.q[1] = *(const uint4*)(row + 16 + half * 8);
  return r.v;
}

// Row reductions across the 16 lanes sharing a C-tile row (xor of lane bits
// 0..3 stays inside each 16-lane half of the wave32).
__device__ __forceinline__ float rowmax16(float x) {
#pragma unroll
  for (int m = 1; m < 16; m <<= 1) x = fmaxf(x, __shfl_xor(x, m, 32));
  return x;
}
__device__ __forceinline__ float rowsum16(float x) {
#pragma unroll
  for (int m = 1; m < 16; m <<= 1) x += __shfl_xor(x, m, 32);
  return x;
}

// ---------------------------------------------------------------------------
// Kernel 0: pack [w_src;w_ref;w_gate] -> bf16 Wt[384][256] (row-major)
// ---------------------------------------------------------------------------
__global__ void pack_weights(const float* __restrict__ wsrc,
                             const float* __restrict__ wref,
                             const float* __restrict__ wgate,
                             unsigned short* __restrict__ Wt) {
  int i = blockIdx.x * blockDim.x + threadIdx.x;
  if (i >= 384 * 256) return;
  int o = i >> 8, c = i & 255;
  float v = (o < 64)  ? wsrc[o * 256 + c]
          : (o < 128) ? wref[(o - 64) * 256 + c]
                      : wgate[(o - 128) * 256 + c];
  Wt[i] = f2bf(v);
}

// ---------------------------------------------------------------------------
// Kernel 1: projections (1x1 convs) via WMMA.
//   Q[b][q][d] (q-major, d contiguous)   <- w_src  x source_features
//   K[b][k][d] (k-major, d contiguous)   <- w_ref  x reference_features
//   G[b][c][k] (c-major, k contiguous)   <- w_gate x reference_features
// grid = B * (HW/64) blocks, 256 threads (8 waves). Wave: 16 queries x 192 outs.
// ---------------------------------------------------------------------------
__global__ void __launch_bounds__(256) proj_kernel(
    const float* __restrict__ srcF, const float* __restrict__ refF,
    const unsigned short* __restrict__ Wt,
    const float* __restrict__ b_src, const float* __restrict__ b_ref,
    const float* __restrict__ b_gate,
    unsigned short* __restrict__ Qb, unsigned short* __restrict__ Kb,
    unsigned short* __restrict__ Gb) {
  const int b     = blockIdx.x >> 6;
  const int qBase = (blockIdx.x & 63) * 64;
  const int lane  = threadIdx.x & 31, wv = threadIdx.x >> 5;
  const int ln    = lane & 15, half = lane >> 4;
  const int qsub  = wv & 3, og = wv >> 2;          // og: output half (0/1)
  const int qRow  = qBase + qsub * 16 + ln;
  const float* sB = srcF + (size_t)b * CC * HW;
  const float* rB = refF + (size_t)b * CC * HW;

  v8f acc[12];
#pragma unroll
  for (int t = 0; t < 12; ++t) acc[t] = (v8f){0.f,0.f,0.f,0.f,0.f,0.f,0.f,0.f};

#pragma unroll
  for (int ck = 0; ck < 8; ++ck) {                 // K loop over C=256, step 32
    const int cb = ck * 32;
    BFrag aS{}; BFrag aR{};
#pragma unroll
    for (int i = 0; i < 16; ++i) {
      int c = cb + ((i < 8) ? (half * 8 + i) : (16 + half * 8 + (i - 8)));
      size_t off = (size_t)c * HW + qRow;          // coalesced across lanes (q)
      aR.u[i] = f2bf(rB[off]);
      if (og == 0) aS.u[i] = f2bf(sB[off]);        // only outputs 0..63 need src
    }
#pragma unroll
    for (int t = 0; t < 12; ++t) {
      const int ot = og * 192 + t * 16;
      v16bf bw = ld_frag(Wt + (size_t)(ot + ln) * 256 + cb, half);
      v16bf a  = (ot < 64) ? aS.v : aR.v;
      acc[t] = __builtin_amdgcn_wmma_f32_16x16x32_bf16(
          false, a, false, bw, (short)0, acc[t], false, false);
    }
  }

#pragma unroll
  for (int t = 0; t < 12; ++t) {
    const int oc = og * 192 + t * 16 + ln;         // output channel (C col)
    float bias = (oc < 64)  ? b_src[oc]
               : (oc < 128) ? b_ref[oc - 64]
                            : b_gate[oc - 128];
#pragma unroll
    for (int j = 0; j < 8; ++j) {                  // C-tile row = j + 8*half
      const int q = qBase + qsub * 16 + j + 8 * half;
      unsigned short bv = f2bf(acc[t][j] + bias);
      if (oc < 64)       Qb[(size_t)(b * HW + q) * CQK + oc] = bv;
      else if (oc < 128) Kb[(size_t)(b * HW + q) * CQK + (oc - 64)] = bv;
      else               Gb[((size_t)b * CC + (oc - 128)) * HW + q] = bv;
    }
  }
}

// ---------------------------------------------------------------------------
// Kernel 2: flash attention + residual.
// grid = B * (HW/64) blocks, 256 threads (8 waves).
// Wave: 16 queries (qsub) x 128 channels (cgrp). 64-key blocks, online softmax.
// K (64x64) and V (256x64) blocks double-buffered in LDS via async-to-LDS DMA;
// rows padded to 72 bf16 (144 B) to rotate LDS banks.
// ---------------------------------------------------------------------------
__global__ void __launch_bounds__(256) flash_attn(
    const unsigned short* __restrict__ Qb, const unsigned short* __restrict__ Kb,
    const unsigned short* __restrict__ Gb, const float* __restrict__ srcF,
    const float* __restrict__ gamma, float* __restrict__ out) {
  __shared__ __align__(16) unsigned short Kl[2][64][72];    //  18 KB
  __shared__ __align__(16) unsigned short Vl[2][256][72];   //  72 KB
  __shared__ __align__(16) unsigned short Plds[8][16][72];  //  18 KB

  const int b     = blockIdx.x >> 6;
  const int qBase = (blockIdx.x & 63) * 64;
  const int tid   = threadIdx.x;
  const int lane  = tid & 31, wv = tid >> 5;
  const int ln    = lane & 15, half = lane >> 4;
  const int qsub  = wv & 3, cBase = (wv >> 2) * 128;
  const unsigned short* Qp = Qb + (size_t)b * HW * CQK;
  const unsigned short* Kp = Kb + (size_t)b * HW * CQK;
  const unsigned short* Gp = Gb + (size_t)b * CC * HW;

  const int qRow = qBase + qsub * 16 + ln;
  const v16bf aQ0 = ld_frag(Qp + (size_t)qRow * CQK, half);        // dims 0..31
  const v16bf aQ1 = ld_frag(Qp + (size_t)qRow * CQK + 32, half);   // dims 32..63

  // Stage one 64-key block: K rows (64x64 bf16) + V rows (256x64 bf16).
  auto stage = [&](int kb, int buf) {
#pragma unroll
    for (int i = 0; i < 2; ++i) {                  // 512 x 16B chunks for K
      int chunk = i * 256 + tid;
      int row = chunk >> 3, col = (chunk & 7) * 8;
      cp16(Kp + (size_t)(kb + row) * CQK + col, &Kl[buf][row][col]);
    }
#pragma unroll
    for (int i = 0; i < 8; ++i) {                  // 2048 x 16B chunks for V
      int chunk = i * 256 + tid;
      int row = chunk >> 3, col = (chunk & 7) * 8;
      cp16(Gp + (size_t)row * HW + kb + col, &Vl[buf][row][col]);
    }
  };

  // softmax in base-2: logits * (1/sqrt(C)) * log2(e)
  const float SCALE = 0.0625f * 1.44269504088896340736f;
  const v8f Z = (v8f){0.f,0.f,0.f,0.f,0.f,0.f,0.f,0.f};

  v8f O[8];
  float mrun[8], lrun[8];
#pragma unroll
  for (int n = 0; n < 8; ++n) O[n] = Z;
#pragma unroll
  for (int j = 0; j < 8; ++j) { mrun[j] = -__builtin_inff(); lrun[j] = 0.f; }

  stage(0, 0);                                     // prologue: block 0 -> buf 0

  for (int ib = 0; ib < HW / 64; ++ib) {           // 64 key-blocks of 64 keys
    const int kb  = ib * 64;
    const int buf = ib & 1;

    wait_async();          // this wave's DMA for `buf` (issued last iter) done
    __syncthreads();       // everyone staged `buf`; prev buffer is dead

    if (ib + 1 < HW / 64) stage(kb + 64, buf ^ 1); // overlap DMA with WMMAs

    float Sv[4][8], mnew[8];
#pragma unroll
    for (int j = 0; j < 8; ++j) mnew[j] = mrun[j];

    // S = Q (16x64) x K^T-block (64x64): 4 key sub-tiles x 2 k-steps
#pragma unroll
    for (int s = 0; s < 4; ++s) {
      const unsigned short* krow = &Kl[buf][s * 16 + ln][0];
      v8f e = Z;
      e = __builtin_amdgcn_wmma_f32_16x16x32_bf16(false, aQ0, false,
              ld_frag(krow, half),      (short)0, e, false, false);
      e = __builtin_amdgcn_wmma_f32_16x16x32_bf16(false, aQ1, false,
              ld_frag(krow + 32, half), (short)0, e, false, false);
#pragma unroll
      for (int j = 0; j < 8; ++j) {
        float v = e[j] * SCALE;
        Sv[s][j] = v;
        mnew[j] = fmaxf(mnew[j], v);
      }
    }
#pragma unroll
    for (int j = 0; j < 8; ++j) mnew[j] = rowmax16(mnew[j]);

    float esc[8], rsum[8];
#pragma unroll
    for (int j = 0; j < 8; ++j) {
      esc[j] = exp2f(mrun[j] - mnew[j]);
      mrun[j] = mnew[j];
      rsum[j] = 0.f;
    }

    // P = exp2(S - m); stash bf16 P row-major in this wave's LDS slab
#pragma unroll
    for (int s = 0; s < 4; ++s)
#pragma unroll
      for (int j = 0; j < 8; ++j) {
        float p = exp2f(Sv[s][j] - mnew[j]);
        rsum[j] += p;
        Plds[wv][j + 8 * half][s * 16 + ln] = f2bf(p);
      }
#pragma unroll
    for (int j = 0; j < 8; ++j) lrun[j] = lrun[j] * esc[j] + rowsum16(rsum[j]);
#pragma unroll
    for (int n = 0; n < 8; ++n)
#pragma unroll
      for (int j = 0; j < 8; ++j) O[n][j] *= esc[j];

    // Reload P in A-layout (same-wave DS ops are in-order)
    const unsigned short* prow = &Plds[wv][ln][0];
    const v16bf aP0 = ld_frag(prow, half);
    const v16bf aP1 = ld_frag(prow + 32, half);

    // O += P (16x64) x V-block (64x128): 8 channel tiles x 2 k-steps
#pragma unroll
    for (int n = 0; n < 8; ++n) {
      const unsigned short* vrow = &Vl[buf][cBase + n * 16 + ln][0];
      O[n] = __builtin_amdgcn_wmma_f32_16x16x32_bf16(false, aP0, false,
                 ld_frag(vrow, half),      (short)0, O[n], false, false);
      O[n] = __builtin_amdgcn_wmma_f32_16x16x32_bf16(false, aP1, false,
                 ld_frag(vrow + 32, half), (short)0, O[n], false, false);
    }
  }

  const float g = gamma[0];
  float inv[8];
#pragma unroll
  for (int j = 0; j < 8; ++j) inv[j] = 1.0f / lrun[j];
#pragma unroll
  for (int n = 0; n < 8; ++n)
#pragma unroll
    for (int j = 0; j < 8; ++j) {
      const int c = cBase + n * 16 + ln;
      const int q = qBase + qsub * 16 + j + 8 * half;
      const size_t idx = ((size_t)b * CC + c) * HW + q;
      out[idx] = g * (O[n][j] * inv[j]) + srcF[idx];
    }
}

// ---------------------------------------------------------------------------
extern "C" void kernel_launch(void* const* d_in, const int* in_sizes, int n_in,
                              void* d_out, int out_size, void* d_ws, size_t ws_size,
                              hipStream_t stream) {
  const float* srcF   = (const float*)d_in[0];
  const float* refF   = (const float*)d_in[1];
  const float* w_src  = (const float*)d_in[2];
  const float* b_src  = (const float*)d_in[3];
  const float* w_ref  = (const float*)d_in[4];
  const float* b_ref  = (const float*)d_in[5];
  const float* w_gate = (const float*)d_in[6];
  const float* b_gate = (const float*)d_in[7];
  const float* gamma  = (const float*)d_in[8];
  float* out = (float*)d_out;

  char* ws = (char*)d_ws;
  unsigned short* Wt = (unsigned short*)(ws);                       // 192 KB
  unsigned short* Qb = (unsigned short*)(ws + 196608);              // 2 MB
  unsigned short* Kb = (unsigned short*)(ws + 196608 + 2097152);    // 2 MB
  unsigned short* Gb = (unsigned short*)(ws + 196608 + 2*2097152);  // 8 MB

  pack_weights<<<384, 256, 0, stream>>>(w_src, w_ref, w_gate, Wt);
  proj_kernel<<<BB * (HW / 64), 256, 0, stream>>>(srcF, refF, Wt,
                                                  b_src, b_ref, b_gate,
                                                  Qb, Kb, Gb);
  flash_attn<<<BB * (HW / 64), 256, 0, stream>>>(Qb, Kb, Gb, srcF, gamma, out);
}
